// CrossAttention_27650999452064
// MI455X (gfx1250) — matrix operand order repaired and checked by reading, not compiled, not run on hardware
//
#include <hip/hip_runtime.h>

typedef __attribute__((ext_vector_type(16))) _Float16 v16h;
typedef __attribute__((ext_vector_type(8)))  float    v8f;
typedef __attribute__((ext_vector_type(4)))  unsigned v4u;
typedef __attribute__((ext_vector_type(4)))  int      v4i;
typedef __attribute__((ext_vector_type(8)))  int      v8i;

#define B_     8
#define CDIM   384
#define QVC    768
#define PCDIM  192
#define HDS    8
#define HC     48
#define HH     128
#define WW     128
#define SDIM   16384L

#if defined(__AMDGCN__) && __has_builtin(__builtin_amdgcn_tensor_load_to_lds)
#define USE_TDM 1
#else
#define USE_TDM 0
#endif

// ---------------------------------------------------------------- f32 -> f16
__global__ __launch_bounds__(256) void cvt_h_kernel(const float* __restrict__ s,
                                                    _Float16* __restrict__ d, long n) {
  long i = (long)blockIdx.x * blockDim.x + threadIdx.x;
  long stride = (long)gridDim.x * blockDim.x;
  for (; i < n; i += stride) d[i] = (_Float16)s[i];
}

__global__ __launch_bounds__(256) void zero_f_kernel(float* __restrict__ p, long n) {
  long i = (long)blockIdx.x * blockDim.x + threadIdx.x;
  if (i < n) p[i] = 0.f;
}

// ------------------------------------------------ 1x1 conv = GEMM via WMMA f16
// Y[b,o,p] = sum_c W[o,c] * X[b,c,p].  W: OxK row-major f16.  X: (B,K,S) f16.
// Tile: BM=64, BN=64, BK=64; 256 threads = 8 waves; wave -> 16x32 C strip;
// A tile staged by the Tensor Data Mover (wave 0), B tile transposed manually.
template <bool OUT_F32>
__global__ __launch_bounds__(256)
void gemm1x1_wmma_kernel(const _Float16* __restrict__ W,
                         const _Float16* __restrict__ X,
                         _Float16* __restrict__ Yh, float* __restrict__ Yf,
                         int O, int K, long S) {
  constexpr int BM = 64, BN = 64, BK = 64;
  __shared__ _Float16 As[BM][BK];      // [m][k], rows contiguous (TDM layout)
  __shared__ _Float16 Bs[BN][BK + 2];  // [n][k] transposed, padded

  const int tid  = threadIdx.x;
  const int lane = tid & 31;
  const int half = lane >> 4;
  const int l16  = lane & 15;
  const int wv   = tid >> 5;
  const int wm   = (wv & 3) * 16;   // wave M offset in tile
  const int wn   = (wv >> 2) * 32;  // wave N offset in tile

  const int  m0 = blockIdx.y * BM;
  const long n0 = (long)blockIdx.x * BN;
  const int  b  = blockIdx.z;
  const _Float16* Xb = X + (size_t)b * K * S;

  v8f acc0 = {}; v8f acc1 = {};

  for (int k0 = 0; k0 < K; k0 += BK) {
#if USE_TDM
    // --- A tile via Tensor Data Mover: BM x BK 2-byte tile, row stride K.
    if (tid < 32) {
      unsigned long long ga =
          (unsigned long long)(const void*)(W + (size_t)m0 * K + k0);
      unsigned ldsa = (unsigned)(unsigned long long)&As[0][0];
      v4u g0;
      g0[0] = 1u;                                   // count=1 (valid descriptor)
      g0[1] = ldsa;                                 // lds_addr (bytes)
      g0[2] = (unsigned)ga;                         // global_addr[31:0]
      g0[3] = (unsigned)((ga >> 32) & 0x01FFFFFFu)  // global_addr[56:32]
              | (2u << 30);                         // type = 2 ("image")
      v8i g1;
      g1[0] = 1 << 16;                                        // data_size=2B
      g1[1] = (int)(((unsigned)K & 0xffffu) << 16);           // tensor_dim0 lo
      g1[2] = (int)(((unsigned)K >> 16) |                     // tensor_dim0 hi
                    (((unsigned)O & 0xffffu) << 16));         // tensor_dim1 lo
      g1[3] = (int)((unsigned)BK << 16);                      // tile_dim0
      g1[4] = BM;                                             // tile_dim1
      g1[5] = K;                                              // dim0_stride lo
      g1[6] = 0;
      g1[7] = 0;
      v4i z4 = {0, 0, 0, 0};
#if __clang_major__ >= 23
      v8i z8 = {0, 0, 0, 0, 0, 0, 0, 0};
      __builtin_amdgcn_tensor_load_to_lds(g0, g1, z4, z4, z8, 0);
#else
      __builtin_amdgcn_tensor_load_to_lds(g0, g1, z4, z4, 0);
#endif
    }
#else
    // --- fallback: manual A tile copy (16 f16 per thread)
    {
      int r  = tid >> 2;
      int cg = (tid & 3) * 16;
      const _Float16* src = W + (size_t)(m0 + r) * K + k0 + cg;
      *(float4*)&As[r][cg]     = *(const float4*)src;
      *(float4*)&As[r][cg + 8] = *(const float4*)(src + 8);
    }
#endif
    // --- B tile (transposed into [n][k]): 16 f16 per thread, contiguous global
    {
      int kk = tid >> 3;
      int nn = (tid & 7) * 8;
#pragma unroll
      for (int kh = 0; kh < 2; ++kh) {
        const _Float16* bsrc = Xb + (size_t)(k0 + kk + kh * 32) * S + n0 + nn;
        if (kh == 0 && k0 + BK < K)
          __builtin_prefetch(bsrc + (size_t)BK * S, 0, 1);  // global_prefetch
        _Float16 tmp[8];
        *(float4*)tmp = *(const float4*)bsrc;
#pragma unroll
        for (int j = 0; j < 8; ++j) Bs[nn + j][kk + kh * 32] = tmp[j];
      }
    }
#if USE_TDM
    if (tid < 32) __builtin_amdgcn_s_wait_tensorcnt(0);
#endif
    __syncthreads();

    // --- two K=32 WMMA steps per stage
#pragma unroll
    for (int ks = 0; ks < BK; ks += 32) {
      v16h a;
      {
        const int m = wm + l16;
#pragma unroll
        for (int p = 0; p < 8; ++p) {
          int kb = ks + ((p >= 4) ? 16 : 0) + half * 8 + 2 * (p & 3);
          a[2 * p]     = As[m][kb];
          a[2 * p + 1] = As[m][kb + 1];
        }
      }
      v16h b0, b1;
      {
        const int n = wn + l16;
#pragma unroll
        for (int i = 0; i < 16; ++i) {
          b0[i] = Bs[n][ks + ((i >= 8) ? 16 : 0) + half * 8 + (i & 7)];
          b1[i] = Bs[n + 16][ks + ((i >= 8) ? 16 : 0) + half * 8 + (i & 7)];
        }
      }
      acc0 = __builtin_amdgcn_wmma_f32_16x16x32_f16(false, a, false, b0, (short)0,
                                                    acc0, false, false);
      acc1 = __builtin_amdgcn_wmma_f32_16x16x32_f16(false, a, false, b1, (short)0,
                                                    acc1, false, false);
    }
    __syncthreads();
  }

  // --- store C (16x16 f32 layout: lane cols, VGPR r -> M = r + 8*(lane>=16))
  const int  cm = m0 + wm + half * 8;
  const long cn = n0 + wn + l16;
#pragma unroll
  for (int r = 0; r < 8; ++r) {
    size_t idx = ((size_t)b * O + cm + r) * S + cn;
    if (OUT_F32) {
      Yf[idx]      = acc0[r];
      Yf[idx + 16] = acc1[r];
    } else {
      Yh[idx]      = (_Float16)acc0[r];
      Yh[idx + 16] = (_Float16)acc1[r];
    }
  }
}

// ---------------------------------------------------------- depthwise 3x3 SAME
__global__ __launch_bounds__(256)
void dwconv3x3_kernel(const _Float16* __restrict__ in, const float* __restrict__ w,
                      _Float16* __restrict__ out, int C) {
  const int b = blockIdx.z, c = blockIdx.y;
  const int p = blockIdx.x * blockDim.x + threadIdx.x;
  const int y = p >> 7, x = p & 127;
  const _Float16* ip = in + ((size_t)b * C + c) * SDIM;
  const float* wc = w + (size_t)c * 9;
  float s = 0.f;
#pragma unroll
  for (int dy = -1; dy <= 1; ++dy)
#pragma unroll
    for (int dx = -1; dx <= 1; ++dx) {
      int yy = y + dy, xx = x + dx;
      if (yy >= 0 && yy < HH && xx >= 0 && xx < WW)
        s += wc[(dy + 1) * 3 + (dx + 1)] * (float)ip[yy * WW + xx];
    }
  out[((size_t)b * C + c) * SDIM + p] = (_Float16)s;
}

// ------------------------------------------------- 1/max(||row||, eps) over S
__global__ __launch_bounds__(256)
void rownorm_kernel(const _Float16* __restrict__ t, float* __restrict__ inv,
                    int Ctot, int Cuse) {
  const int r  = blockIdx.x;  // 0 .. B*Cuse-1
  const int b  = r / Cuse;
  const int ch = r % Cuse;
  const _Float16* p = t + ((size_t)b * Ctot + ch) * SDIM;
  float s = 0.f;
  for (long i = threadIdx.x; i < SDIM; i += 256) {
    float v = (float)p[i];
    s += v * v;
  }
  __shared__ float red[256];
  red[threadIdx.x] = s;
  __syncthreads();
  for (int o = 128; o > 0; o >>= 1) {
    if (threadIdx.x < o) red[threadIdx.x] += red[threadIdx.x + o];
    __syncthreads();
  }
  if (threadIdx.x == 0) inv[r] = 1.0f / fmaxf(sqrtf(red[0]), 1e-12f);
}

// --------------------------------------- attn_raw[bh,c,d] += sum_s q*k (split s)
__global__ __launch_bounds__(256)
void attn_qk_kernel(const _Float16* __restrict__ q, const _Float16* __restrict__ k,
                    float* __restrict__ attn) {
  const int bh = blockIdx.y;
  const int b = bh >> 3, h = bh & 7;
  __shared__ _Float16 qs[HC][64 + 2];
  __shared__ _Float16 ks[HC][64 + 2];
  const _Float16* qp = q + ((size_t)b * QVC + h * HC) * SDIM;  // q = qv ch 0..383
  const _Float16* kp = k + ((size_t)b * CDIM + h * HC) * SDIM;

  float acc[9] = {};
  const long span = SDIM / gridDim.x;
  const long sb   = (long)blockIdx.x * span;
  for (long s0 = sb; s0 < sb + span; s0 += 64) {
    for (int e = threadIdx.x; e < HC * 64; e += 256) {
      int cc = e >> 6, ss = e & 63;
      qs[cc][ss] = qp[(size_t)cc * SDIM + s0 + ss];
      ks[cc][ss] = kp[(size_t)cc * SDIM + s0 + ss];
    }
    __syncthreads();
#pragma unroll
    for (int j = 0; j < 9; ++j) {
      int pr = threadIdx.x + j * 256;  // 2304 (c,d) pairs
      int c = pr / HC, d = pr % HC;
      float s = 0.f;
      for (int ss = 0; ss < 64; ++ss) s += (float)qs[c][ss] * (float)ks[d][ss];
      acc[j] += s;
    }
    __syncthreads();
  }
#pragma unroll
  for (int j = 0; j < 9; ++j) {
    int pr = threadIdx.x + j * 256;
    atomicAdd(&attn[(size_t)bh * HC * HC + pr], acc[j]);
  }
}

// -------------------- softmax over d with temperature and 1/|q| 1/|k| folding
__global__ __launch_bounds__(64)
void softmax48_kernel(float* __restrict__ attn, const float* __restrict__ invq,
                      const float* __restrict__ invk, const float* __restrict__ temp) {
  const int c = blockIdx.x, bh = blockIdx.y;
  const int b = bh >> 3, h = bh & 7;
  float* row = attn + (size_t)bh * HC * HC + (size_t)c * HC;
  const int d = threadIdx.x;
  const float scl = invq[b * CDIM + h * HC + c] * temp[h];
  float val = 0.f, m = -1e30f;
  if (d < HC) {
    val = row[d] * scl * invk[b * CDIM + h * HC + d];
    m = val;
  }
  __shared__ float red[64];
  red[d] = m;
  __syncthreads();
  for (int o = 32; o > 0; o >>= 1) {
    if (d < o) red[d] = fmaxf(red[d], red[d + o]);
    __syncthreads();
  }
  const float mx = red[0];
  __syncthreads();
  float e = (d < HC) ? expf(val - mx) : 0.f;
  red[d] = e;
  __syncthreads();
  for (int o = 32; o > 0; o >>= 1) {
    if (d < o) red[d] += red[d + o];
    __syncthreads();
  }
  if (d < HC) row[d] = e / red[0];
}

// ------------------------------------------- out[c,s] = sum_d attn[c,d]*v[d,s]
__global__ __launch_bounds__(256)
void attn_v_kernel(const float* __restrict__ attn, const _Float16* __restrict__ qv,
                   _Float16* __restrict__ out) {
  const int bh = blockIdx.y;
  const int b = bh >> 3, h = bh & 7;
  const long s0 = (long)blockIdx.x * 128;
  __shared__ float    a_s[HC][HC];
  __shared__ _Float16 vs[HC][128 + 4];
  for (int e = threadIdx.x; e < HC * HC; e += 256)
    a_s[e / HC][e % HC] = attn[(size_t)bh * HC * HC + e];
  const _Float16* vp = qv + ((size_t)b * QVC + CDIM + h * HC) * SDIM;  // v = ch 384..767
  for (int e = threadIdx.x; e < HC * 128; e += 256) {
    int dd = e >> 7, ss = e & 127;
    vs[dd][ss] = vp[(size_t)dd * SDIM + s0 + ss];
  }
  __syncthreads();
  for (int e = threadIdx.x; e < HC * 128; e += 256) {
    int c = e >> 7, ss = e & 127;
    float s = 0.f;
#pragma unroll
    for (int dd = 0; dd < HC; ++dd) s += a_s[c][dd] * (float)vs[dd][ss];
    out[((size_t)b * CDIM + h * HC + c) * SDIM + s0 + ss] = (_Float16)s;
  }
}

// -----------------------------------------------------------------------------
extern "C" void kernel_launch(void* const* d_in, const int* in_sizes, int n_in,
                              void* d_out, int out_size, void* d_ws, size_t ws_size,
                              hipStream_t stream) {
  const float* x      = (const float*)d_in[0];
  const float* de     = (const float*)d_in[1];
  const float* w_qv   = (const float*)d_in[2];
  const float* w_qvdw = (const float*)d_in[3];
  const float* w_k    = (const float*)d_in[4];
  const float* w_kdw  = (const float*)d_in[5];
  const float* w_proj = (const float*)d_in[6];
  const float* temp   = (const float*)d_in[7];
  float* out = (float*)d_out;

  char* ws = (char*)d_ws;
  size_t off = 0;
  auto alloc_h = [&](size_t n) { _Float16* p = (_Float16*)(ws + off); off += ((n * 2 + 255) / 256) * 256; return p; };
  auto alloc_f = [&](size_t n) { float* p = (float*)(ws + off); off += ((n * 4 + 255) / 256) * 256; return p; };

  _Float16* xh     = alloc_h((size_t)B_ * CDIM * SDIM);     // 100.7 MB (reused as aout)
  _Float16* deh    = alloc_h((size_t)B_ * PCDIM * SDIM);    // 50.3 MB
  _Float16* wqvh   = alloc_h((size_t)QVC * CDIM);
  _Float16* wkh    = alloc_h((size_t)CDIM * PCDIM);
  _Float16* wprojh = alloc_h((size_t)CDIM * CDIM);
  _Float16* qvdw   = alloc_h((size_t)B_ * QVC * SDIM);      // 201 MB
  _Float16* qvpre  = alloc_h((size_t)B_ * QVC * SDIM);      // 201 MB (reused for k)
  float* invq = alloc_f((size_t)B_ * CDIM);
  float* invk = alloc_f((size_t)B_ * CDIM);
  float* attn = alloc_f((size_t)B_ * HDS * HC * HC);
  _Float16* kpre = qvpre;                                   // reuse qvpre region
  _Float16* kdw  = qvpre + (size_t)B_ * CDIM * SDIM;
  _Float16* aout = xh;                                      // reuse x region

  // 1. f32 -> f16 conversions
  cvt_h_kernel<<<2048, 256, 0, stream>>>(x, xh, (long)B_ * CDIM * SDIM);
  cvt_h_kernel<<<2048, 256, 0, stream>>>(de, deh, (long)B_ * PCDIM * SDIM);
  cvt_h_kernel<<<256, 256, 0, stream>>>(w_qv, wqvh, (long)QVC * CDIM);
  cvt_h_kernel<<<128, 256, 0, stream>>>(w_k, wkh, (long)CDIM * PCDIM);
  cvt_h_kernel<<<128, 256, 0, stream>>>(w_proj, wprojh, (long)CDIM * CDIM);

  // 2. qv = conv1x1(x, w_qv)   [WMMA + TDM]
  gemm1x1_wmma_kernel<false><<<dim3(SDIM / 64, QVC / 64, B_), 256, 0, stream>>>(
      wqvh, xh, qvpre, nullptr, QVC, CDIM, SDIM);
  // 3. depthwise 3x3 on qv
  dwconv3x3_kernel<<<dim3(SDIM / 256, QVC, B_), 256, 0, stream>>>(qvpre, w_qvdw, qvdw, QVC);

  // 4. k = conv1x1(de, w_k)    [WMMA + TDM]
  gemm1x1_wmma_kernel<false><<<dim3(SDIM / 64, CDIM / 64, B_), 256, 0, stream>>>(
      wkh, deh, kpre, nullptr, CDIM, PCDIM, SDIM);
  // 5. depthwise 3x3 on k
  dwconv3x3_kernel<<<dim3(SDIM / 256, CDIM, B_), 256, 0, stream>>>(kpre, w_kdw, kdw, CDIM);

  // 6. row norms for q (first 384 ch of qvdw) and k
  rownorm_kernel<<<B_ * CDIM, 256, 0, stream>>>(qvdw, invq, QVC, CDIM);
  rownorm_kernel<<<B_ * CDIM, 256, 0, stream>>>(kdw, invk, CDIM, CDIM);

  // 7. attention Gram matrix (raw), split over s with atomic accumulation
  zero_f_kernel<<<(B_ * HDS * HC * HC + 255) / 256, 256, 0, stream>>>(
      attn, (long)B_ * HDS * HC * HC);
  attn_qk_kernel<<<dim3(16, B_ * HDS), 256, 0, stream>>>(qvdw, kdw, attn);

  // 8. softmax with temperature + norm folding
  softmax48_kernel<<<dim3(HC, B_ * HDS), 64, 0, stream>>>(attn, invq, invk, temp);

  // 9. out = attn @ v
  attn_v_kernel<<<dim3(SDIM / 128, B_ * HDS), 256, 0, stream>>>(attn, qvdw, aout);

  // 10. final projection conv1x1 -> f32 d_out   [WMMA + TDM]
  gemm1x1_wmma_kernel<true><<<dim3(SDIM / 64, CDIM / 64, B_), 256, 0, stream>>>(
      wprojh, aout, nullptr, out, CDIM, CDIM, SDIM);
}
